// Head_1915555414232
// MI455X (gfx1250) — matrix operand order repaired and checked by reading, not compiled, not run on hardware
//
#include <hip/hip_runtime.h>
#include <hip/hip_bf16.h>

// ---------------------------------------------------------------------------
// Fused causal single-head attention for MI455X (gfx1250, wave32, WMMA).
//   B=256, T=256, C=384, H=64.  out = softmax(mask(QK^T * H^-0.5)) @ V
// Kernel 0: convert Wk/Wq/Wv to f16 once (removes per-block re-conversion).
// Kernel 1: q/k/v = x @ W^T. x tile staged f32->f16 into LDS cooperatively
//           (bulk loads, one barrier), A-frags via ds_load_b128, W-frags via
//           global_load_b128 of f16 (L2-resident). f16 WMMA, f32 accum.
// Kernel 2: flash-style online-softmax attention, V^T staged in LDS,
//           P transposed D-layout -> A-layout through per-wave LDS scratch.
// ---------------------------------------------------------------------------

typedef _Float16 v16h __attribute__((ext_vector_type(16)));
typedef _Float16 v8h  __attribute__((ext_vector_type(8)));
typedef float    v8f  __attribute__((ext_vector_type(8)));

constexpr int Bn = 256;   // batch
constexpr int Tn = 256;   // sequence
constexpr int Cn = 384;   // model dim
constexpr int Hn = 64;    // head dim
constexpr int XS_STRIDE = 392;      // 384 + 8 pad: 196 dwords/row, 196%64=4
constexpr float SCALE   = 0.125f;   // 64^-0.5
constexpr float NEG_BIG = -1e30f;

static __device__ __forceinline__ v8f wmma16(v16h a, v16h b, v8f c) {
  return __builtin_amdgcn_wmma_f32_16x16x32_f16(false, a, false, b,
                                                (short)0, c, false, false);
}

static __device__ __forceinline__ v8h cvt8(float4 f0, float4 f1) {
  v8h r;
  r[0]=(_Float16)f0.x; r[1]=(_Float16)f0.y; r[2]=(_Float16)f0.z; r[3]=(_Float16)f0.w;
  r[4]=(_Float16)f1.x; r[5]=(_Float16)f1.y; r[6]=(_Float16)f1.z; r[7]=(_Float16)f1.w;
  return r;
}

// Load two runs of 8 consecutive f16 (16B each) into one v16h fragment.
static __device__ __forceinline__ v16h ld2x8h(const _Float16* p0,
                                              const _Float16* p1) {
  v8h a = *reinterpret_cast<const v8h*>(p0);
  v8h b = *reinterpret_cast<const v8h*>(p1);
  return __builtin_shufflevector(a, b, 0,1,2,3,4,5,6,7,8,9,10,11,12,13,14,15);
}

// ---------------------------------------------------------------------------
// Kernel 0: one-shot f32 -> f16 conversion of the three weight matrices.
// wf layout: [3][Hn*Cn], order Wk, Wq, Wv.  grid: 36 blocks x 256 threads.
// ---------------------------------------------------------------------------
__global__ __launch_bounds__(256)
void cvt_weights(const float* __restrict__ Wk, const float* __restrict__ Wq,
                 const float* __restrict__ Wv, _Float16* __restrict__ wf) {
  const int idx = blockIdx.x * 256 + threadIdx.x;   // one v8h per thread
  const int per = (Hn * Cn) / 8;                    // 3072 chunks per matrix
  const int m   = idx / per;
  const int e   = (idx - m * per) * 8;
  const float* src = (m == 0) ? Wk : (m == 1) ? Wq : Wv;
  const float4* s4 = reinterpret_cast<const float4*>(src + e);
  float4 f0 = s4[0], f1 = s4[1];
  *reinterpret_cast<v8h*>(wf + (size_t)m * (Hn * Cn) + e) = cvt8(f0, f1);
}

// ---------------------------------------------------------------------------
// Kernel 1: projections.  grid = B*(T/16) blocks, 128 threads (4 waves).
// Wave w computes out cols [16w,16w+16) of K,Q,V for one 16-row x-tile.
// ---------------------------------------------------------------------------
__global__ __launch_bounds__(128)
void proj_qkv(const float* __restrict__ x,
              const _Float16* __restrict__ wf,
              _Float16* __restrict__ qws,
              _Float16* __restrict__ kws,
              _Float16* __restrict__ vws) {
  __shared__ __align__(16) _Float16 Xs[16 * XS_STRIDE];   // 12.25 KB f16 x-tile

  const int tid  = threadIdx.x;
  const int lane = tid & 31;
  const int wave = tid >> 5;
  const int b    = blockIdx.x >> 4;          // T/16 = 16 row tiles per batch
  const int t0   = (blockIdx.x & 15) << 4;
  const int n0   = wave << 4;

  // ---- cooperative stage: x tile (16x384 f32) -> f16 in LDS -------------
  // 768 v8h chunks; 6 per thread; loads bulk-issued, converts batched after.
  for (int c = tid; c < 16 * (Cn / 8); c += 128) {
    const int row  = c / (Cn / 8);
    const int col8 = (c - row * (Cn / 8)) * 8;
    const float4* p =
        reinterpret_cast<const float4*>(x + (size_t)(b * Tn + t0 + row) * Cn + col8);
    float4 f0 = p[0], f1 = p[1];
    *reinterpret_cast<v8h*>(&Xs[row * XS_STRIDE + col8]) = cvt8(f0, f1);
  }
  __syncthreads();

  // A-frag (contiguous k order, same as B-frag): lane%16 = M, K = [k0+hi .. +15], hi = 0/16
  const int mrow = lane & 15;
  const int hiA  = (lane >> 4) << 4;         // 0 / 16
  // B-frag: lane%16 = N, K = [k0+koff .. +15], koff = 0/16.
  const int ncol = n0 + (lane & 15);
  const int koff = (lane >> 4) << 4;         // 0 / 16

  const _Float16* wkr = wf + (size_t)0 * Hn * Cn + (size_t)ncol * Cn;
  const _Float16* wqr = wf + (size_t)1 * Hn * Cn + (size_t)ncol * Cn;
  const _Float16* wvr = wf + (size_t)2 * Hn * Cn + (size_t)ncol * Cn;
  const _Float16* xr  = &Xs[mrow * XS_STRIDE];

  v8f ck = {}, cq = {}, cv = {};
#pragma unroll
  for (int kt = 0; kt < Cn / 32; ++kt) {
    const int k0 = kt * 32;
    v16h a  = ld2x8h(xr + k0 + hiA, xr + k0 + hiA + 8);           // ds_load_b128 x2
    v16h bk = ld2x8h(wkr + k0 + koff, wkr + k0 + koff + 8);        // global b128 x2
    v16h bq = ld2x8h(wqr + k0 + koff, wqr + k0 + koff + 8);
    v16h bv = ld2x8h(wvr + k0 + koff, wvr + k0 + koff + 8);
    ck = wmma16(a, bk, ck);
    cq = wmma16(a, bq, cq);
    cv = wmma16(a, bv, cv);
  }

  // D layout: VGPR r -> row M = r (+8 for upper lane half), col N = lane%16.
  const int hm = (lane >> 4) << 3;
#pragma unroll
  for (int r = 0; r < 8; ++r) {
    const int t = t0 + r + hm;
    const size_t o = (size_t)(b * Tn + t) * Hn + n0 + (lane & 15);
    kws[o] = (_Float16)ck[r];
    qws[o] = (_Float16)cq[r];
    vws[o] = (_Float16)cv[r];
  }
}

// ---------------------------------------------------------------------------
// Kernel 2: flash attention.  grid = B*(T/64), 128 threads (4 waves).
// Wave w owns rows [band*64 + 16w, +16).  V^T staged in LDS, K/Q from global
// f16 workspace (L2/WGP$-resident), P transposed through per-wave LDS scratch.
// ---------------------------------------------------------------------------
__global__ __launch_bounds__(128)
void attn_fused(const _Float16* __restrict__ qws,
                const _Float16* __restrict__ kws,
                const _Float16* __restrict__ vws,
                float* __restrict__ out) {
  __shared__ __align__(16) _Float16 Vt[Hn * Tn];       // [h][s], 32 KB
  __shared__ __align__(16) _Float16 Psh[4][16 * 40];   // per-wave P scratch, 5 KB

  const int tid  = threadIdx.x;
  const int lane = tid & 31;
  const int wave = tid >> 5;
  const int b    = blockIdx.x >> 2;
  const int band = blockIdx.x & 3;
  const int tt0  = band * 64 + wave * 16;

  // ---- stage V transposed into LDS --------------------------------------
  const _Float16* vb = vws + (size_t)b * Tn * Hn;
  for (int c = tid; c < (Tn * Hn) / 8; c += 128) {
    const int s  = c >> 3;
    const int h0 = (c & 7) << 3;
    v8h vv = *reinterpret_cast<const v8h*>(vb + s * Hn + h0);
#pragma unroll
    for (int e = 0; e < 8; ++e) Vt[(h0 + e) * Tn + s] = vv[e];
  }
  __syncthreads();

  // ---- load Q fragments (held in registers for whole row-band) ----------
  const int mrow = lane & 15;
  const int hiA  = (lane >> 4) << 4;   // A-frag K-offset half
  const int hm   = (lane >> 4) << 3;   // C/D row offset half
  const int scol = lane & 15;          // N index within 16-tile
  const int koffB = (lane >> 4) << 4;  // B-frag K-offset half

  const _Float16* qrow = qws + (size_t)(b * Tn + tt0 + mrow) * Hn;
  const v16h aq0 = ld2x8h(qrow + hiA,      qrow + hiA + 8);   // h = 0..31
  const v16h aq1 = ld2x8h(qrow + 32 + hiA, qrow + 40 + hiA);  // h = 32..63

  const _Float16* kb = kws + (size_t)b * Tn * Hn;

  v8f acc0 = {}, acc1 = {}, acc2 = {}, acc3 = {};
  float mr[8], lr[8];
#pragma unroll
  for (int r = 0; r < 8; ++r) { mr[r] = NEG_BIG; lr[r] = 0.f; }

  const int jmax = (tt0 + 15) >> 5;    // causal bound on 32-wide s-blocks
  for (int j = 0; j <= jmax; ++j) {
    const int s0 = j << 5;

    // ---- S = Q K^T for columns [s0, s0+32) ------------------------------
    const _Float16* klo = kb + (size_t)(s0 + scol) * Hn;
    const _Float16* khi = kb + (size_t)(s0 + 16 + scol) * Hn;
    v8f slo = {}, shi = {};
    slo = wmma16(aq0, ld2x8h(klo + koffB,      klo + koffB + 8),      slo);
    slo = wmma16(aq1, ld2x8h(klo + 32 + koffB, klo + 32 + koffB + 8), slo);
    shi = wmma16(aq0, ld2x8h(khi + koffB,      khi + koffB + 8),      shi);
    shi = wmma16(aq1, ld2x8h(khi + 32 + koffB, khi + 32 + koffB + 8), shi);
    slo = slo * SCALE;
    shi = shi * SCALE;

    // ---- causal mask (partial blocks only) ------------------------------
    if (s0 + 31 > tt0) {
#pragma unroll
      for (int r = 0; r < 8; ++r) {
        const int t = tt0 + r + hm;
        if (s0 + scol      > t) slo[r] = NEG_BIG;
        if (s0 + 16 + scol > t) shi[r] = NEG_BIG;
      }
    }

    // ---- online softmax (row reductions across 16-lane halves) ----------
#pragma unroll
    for (int r = 0; r < 8; ++r) {
      float rm = fmaxf(slo[r], shi[r]);
      rm = fmaxf(rm, __shfl_xor(rm, 1, 32));
      rm = fmaxf(rm, __shfl_xor(rm, 2, 32));
      rm = fmaxf(rm, __shfl_xor(rm, 4, 32));
      rm = fmaxf(rm, __shfl_xor(rm, 8, 32));
      const float mnew = fmaxf(mr[r], rm);
      const float corr = __expf(mr[r] - mnew);
      const float plo  = __expf(slo[r] - mnew);
      const float phi  = __expf(shi[r] - mnew);
      slo[r] = plo;
      shi[r] = phi;
      float rs = plo + phi;
      rs += __shfl_xor(rs, 1, 32);
      rs += __shfl_xor(rs, 2, 32);
      rs += __shfl_xor(rs, 4, 32);
      rs += __shfl_xor(rs, 8, 32);
      lr[r] = lr[r] * corr + rs;
      mr[r] = mnew;
      acc0[r] *= corr; acc1[r] *= corr; acc2[r] *= corr; acc3[r] *= corr;
    }

    // ---- transpose P: D-layout -> A-layout via per-wave LDS scratch -----
    _Float16* pw = &Psh[wave][0];
#pragma unroll
    for (int r = 0; r < 8; ++r) {
      const int row = r + hm;
      pw[row * 40 + scol]      = (_Float16)slo[r];
      pw[row * 40 + 16 + scol] = (_Float16)shi[r];
    }
    // same-wave LDS RAW: LDS ops are in-order per wave; wait + fence anyway.
    asm volatile("s_wait_dscnt 0" ::: "memory");
    const _Float16* prow = pw + mrow * 40;
    const v16h ap = ld2x8h(prow + hiA, prow + hiA + 8);

    // ---- acc += P @ V  (4 col tiles of 16 over H=64) --------------------
    {
      const _Float16* v0 = &Vt[(0  + scol) * Tn + s0 + koffB];
      const _Float16* v1 = &Vt[(16 + scol) * Tn + s0 + koffB];
      const _Float16* v2 = &Vt[(32 + scol) * Tn + s0 + koffB];
      const _Float16* v3 = &Vt[(48 + scol) * Tn + s0 + koffB];
      acc0 = wmma16(ap, ld2x8h(v0, v0 + 8), acc0);
      acc1 = wmma16(ap, ld2x8h(v1, v1 + 8), acc1);
      acc2 = wmma16(ap, ld2x8h(v2, v2 + 8), acc2);
      acc3 = wmma16(ap, ld2x8h(v3, v3 + 8), acc3);
    }
  }

  // ---- epilogue: divide by row sum, store fp32 --------------------------
#pragma unroll
  for (int r = 0; r < 8; ++r) {
    const int t = tt0 + r + hm;
    const float inv = 1.0f / lr[r];
    float* orow = out + (size_t)(b * Tn + t) * Hn + scol;
    orow[0]  = acc0[r] * inv;
    orow[16] = acc1[r] * inv;
    orow[32] = acc2[r] * inv;
    orow[48] = acc3[r] * inv;
  }
}

// ---------------------------------------------------------------------------
extern "C" void kernel_launch(void* const* d_in, const int* in_sizes, int n_in,
                              void* d_out, int out_size, void* d_ws, size_t ws_size,
                              hipStream_t stream) {
  (void)in_sizes; (void)n_in; (void)out_size; (void)ws_size;
  const float* x  = (const float*)d_in[0];
  const float* Wk = (const float*)d_in[1];
  const float* Wq = (const float*)d_in[2];
  const float* Wv = (const float*)d_in[3];
  float* out = (float*)d_out;

  const size_t elems = (size_t)Bn * Tn * Hn;          // 4.19M f16 each
  _Float16* qws = reinterpret_cast<_Float16*>(d_ws);  // 3*elems*2 = 25.2 MB
  _Float16* kws = qws + elems;
  _Float16* vws = kws + elems;
  _Float16* wf  = vws + elems;                        // + 3*64*384*2 = 147 KB

  cvt_weights<<<dim3((3 * Hn * Cn / 8) / 256), 256, 0, stream>>>(Wk, Wq, Wv, wf);
  proj_qkv<<<dim3(Bn * (Tn / 16)), 128, 0, stream>>>(x, wf, qws, kws, vws);
  attn_fused<<<dim3(Bn * (Tn / 64)), 128, 0, stream>>>(qws, kws, vws, out);
}
